// GraspEvaluator_67190468378989
// MI455X (gfx1250) — matrix operand order repaired and checked by reading, not compiled, not run on hardware
//
#include <hip/hip_runtime.h>

typedef unsigned short u16;
typedef __attribute__((ext_vector_type(16))) __bf16 v16bf;
typedef __attribute__((ext_vector_type(8)))  float  v8f;

#define BN_EPS 1e-5f

__device__ __forceinline__ u16 f2bf(float f) {
    unsigned u = __float_as_uint(f);
    u += 0x7FFFu + ((u >> 16) & 1u);        // round-to-nearest-even
    return (u16)(u >> 16);
}
__device__ __forceinline__ unsigned f2bf_pk(float lo, float hi) {
    return (unsigned)f2bf(lo) | ((unsigned)f2bf(hi) << 16);
}
__device__ __forceinline__ float bf2f(u16 h) {
    return __uint_as_float(((unsigned)h) << 16);
}

union Frag {
    uint4 q[2];
    u16   u[16];
    v16bf v;
};

// ---------------------------------------------------------------------------
// build xyz0 = concat(pc, gripper_pc) along point dim  (B,1280,3) fp32
// ---------------------------------------------------------------------------
__global__ void build_xyz_kernel(const float* __restrict__ pc,
                                 const float* __restrict__ gp,
                                 float* __restrict__ xyz,
                                 int B, int NP, int NG) {
    int i = blockIdx.x * blockDim.x + threadIdx.x;
    int N = NP + NG;
    int total = B * N * 3;
    if (i >= total) return;
    int d = i % 3;
    int n = (i / 3) % N;
    int b = i / (3 * N);
    xyz[i] = (n < NP) ? pc[(size_t)(b * NP + n) * 3 + d]
                      : gp[(size_t)(b * NG + (n - NP)) * 3 + d];
}

// ---------------------------------------------------------------------------
// One-time weight prep: fp32 (Kreal x N) -> bf16, zero-padded to Kpad, stored
// pre-swizzled per (32K x 64N) tile in the exact LDS / B-fragment image order:
//   tile t = (k/32, n/64); within tile, element (kk, nn) lives at
//   (ct*32 + (kk>>4)*16 + col)*16 + (kk&15)  with ct=nn>>4, col=nn&15.
// GEMM staging then becomes a raw tile copy (no predicates, no converts).
// ---------------------------------------------------------------------------
__global__ void prep_w_kernel(const float* __restrict__ W, u16* __restrict__ Wsw,
                              int N, int Kreal, int Kpad) {
    int i = blockIdx.x * blockDim.x + threadIdx.x;
    if (i >= Kpad * N) return;
    int k = i / N, n = i % N;
    float w = (k < Kreal) ? W[(size_t)k * N + n] : 0.f;
    int kb = k >> 5, kk = k & 31;
    int nb = n >> 6, nn = n & 63;
    int ct = nn >> 4, col = nn & 15;
    int dl = ((kk >> 4) << 4) + col;
    Wsw[((size_t)(kb * (N >> 6) + nb)) * 2048 + (ct * 32 + dl) * 16 + (kk & 15)] = f2bf(w);
}

// ---------------------------------------------------------------------------
// Farthest-point sampling: one block per batch, LDS distance array + argmax
// Writes new_xyz (B,npoint,3). Matches reference: idx[0]=0 then argmax chain.
// ---------------------------------------------------------------------------
__global__ __launch_bounds__(256)
void fps_kernel(const float* __restrict__ xyz, float* __restrict__ new_xyz,
                int N, int npoint) {
    __shared__ float dist[1280];
    __shared__ float rv[256];
    __shared__ int   ri[256];
    __shared__ int   s_last;
    const int b = blockIdx.x, tid = threadIdx.x;
    const float* X = xyz + (size_t)b * N * 3;
    for (int i = tid; i < N; i += 256) dist[i] = 1.0e10f;
    int last = 0;
    for (int it = 0; it < npoint; ++it) {
        float lx = X[last * 3 + 0], ly = X[last * 3 + 1], lz = X[last * 3 + 2];
        if (tid == 0) {
            float* o = new_xyz + ((size_t)b * npoint + it) * 3;
            o[0] = lx; o[1] = ly; o[2] = lz;
        }
        __syncthreads();
        float bv = -1.0f; int bi = 0;
        for (int i = tid; i < N; i += 256) {
            float dx = X[i * 3 + 0] - lx;
            float dy = X[i * 3 + 1] - ly;
            float dz = X[i * 3 + 2] - lz;
            float nd = fminf(dist[i], dx * dx + dy * dy + dz * dz);
            dist[i] = nd;
            if (nd > bv) { bv = nd; bi = i; }
        }
        rv[tid] = bv; ri[tid] = bi;
        __syncthreads();
        for (int s = 128; s > 0; s >>= 1) {
            if (tid < s) {
                if (rv[tid + s] > rv[tid] ||
                    (rv[tid + s] == rv[tid] && ri[tid + s] < ri[tid])) {
                    rv[tid] = rv[tid + s]; ri[tid] = ri[tid + s];
                }
            }
            __syncthreads();
        }
        if (tid == 0) s_last = ri[0];
        __syncthreads();
        last = s_last;
    }
}

// ---------------------------------------------------------------------------
// SA1 ball-query + grouping: one wave per center. wave32 ballot compaction
// emits neighbors in ascending index order (== reference sorted-candidates).
// out: (B*128, 64, 32) bf16, ch = [gxyz(3), xyz(3), label(1), pad(25)=0]
// ---------------------------------------------------------------------------
__global__ __launch_bounds__(256)
void group1_kernel(const float* __restrict__ xyz,   // (B,1280,3)
                   const float* __restrict__ ctr,   // (B*128,3)
                   u16* __restrict__ out) {
    const int lane = threadIdx.x & 31;
    const int gid  = blockIdx.x * 8 + (threadIdx.x >> 5);   // b*128+s
    const int b    = gid >> 7;
    const int N    = 1280;
    const float R2 = 0.02f * 0.02f;
    const float* X = xyz + (size_t)b * N * 3;
    const float cx = ctr[(size_t)gid * 3 + 0];
    const float cy = ctr[(size_t)gid * 3 + 1];
    const float cz = ctr[(size_t)gid * 3 + 2];
    u16* base = out + (size_t)gid * 64 * 32;

    int count = 0, first = -1;
    for (int b0 = 0; b0 < N && count < 64; b0 += 32) {
        int n = b0 + lane;
        float px = X[n * 3 + 0], py = X[n * 3 + 1], pz = X[n * 3 + 2];
        float dx = px - cx, dy = py - cy, dz = pz - cz;
        bool pred = (dx * dx + dy * dy + dz * dz) < R2;
        unsigned mask = (unsigned)__ballot(pred);
        int pos = count + __popc(mask & ((1u << lane) - 1u));
        if (pred && pos < 64) {
            uint4* d = (uint4*)(base + (size_t)pos * 32);
            d[0] = make_uint4(f2bf_pk(dx, dy), f2bf_pk(dz, px),
                              f2bf_pk(py, pz), f2bf_pk(n < 1024 ? 1.0f : 0.0f, 0.0f));
            d[1] = make_uint4(0, 0, 0, 0);
            d[2] = make_uint4(0, 0, 0, 0);
            d[3] = make_uint4(0, 0, 0, 0);
        }
        if (first < 0 && mask) first = b0 + __ffs(mask) - 1;
        count += __popc(mask);
    }
    int filled = count < 64 ? count : 64;
    int fi = first >= 0 ? first : 0;
    float px = X[fi * 3 + 0], py = X[fi * 3 + 1], pz = X[fi * 3 + 2];
    uint4 q0 = make_uint4(f2bf_pk(px - cx, py - cy), f2bf_pk(pz - cz, px),
                          f2bf_pk(py, pz), f2bf_pk(fi < 1024 ? 1.0f : 0.0f, 0.0f));
    for (int pos = filled + lane; pos < 64; pos += 32) {
        uint4* d = (uint4*)(base + (size_t)pos * 32);
        d[0] = q0;
        d[1] = make_uint4(0, 0, 0, 0);
        d[2] = make_uint4(0, 0, 0, 0);
        d[3] = make_uint4(0, 0, 0, 0);
    }
}

// ---------------------------------------------------------------------------
// SA2 grouping: points = SA1 centers (B,128,3), feat = pooled1 (B,128,128) bf16
// out: (B*32, 128, 160) bf16 = [gxyz(3), feat(128), pad(29)=0]
// ---------------------------------------------------------------------------
__global__ __launch_bounds__(256)
void group2_kernel(const float* __restrict__ xyz,   // (B,128,3)
                   const float* __restrict__ ctr,   // (B*32,3)
                   const u16* __restrict__ feat,    // (B,128,128)
                   u16* __restrict__ out) {
    const int lane = threadIdx.x & 31;
    const int gid  = blockIdx.x * 8 + (threadIdx.x >> 5);   // b*32+s, 2048 total
    const int b    = gid >> 5;
    const int N = 128, NS = 128, CP = 160;
    const float R2 = 0.04f * 0.04f;
    const float* X = xyz + (size_t)b * N * 3;
    const u16*   F = feat + (size_t)b * N * 128;
    const float cx = ctr[(size_t)gid * 3 + 0];
    const float cy = ctr[(size_t)gid * 3 + 1];
    const float cz = ctr[(size_t)gid * 3 + 2];
    u16* base = out + (size_t)gid * NS * CP;

    int count = 0, first = -1;
    for (int b0 = 0; b0 < N && count < NS; b0 += 32) {
        int n = b0 + lane;
        float dx = X[n * 3 + 0] - cx;
        float dy = X[n * 3 + 1] - cy;
        float dz = X[n * 3 + 2] - cz;
        bool pred = (dx * dx + dy * dy + dz * dz) < R2;
        unsigned mask = (unsigned)__ballot(pred);
        int pos = count + __popc(mask & ((1u << lane) - 1u));
        if (pred && pos < NS) {
            u16* d = base + (size_t)pos * CP;
            d[0] = f2bf(dx); d[1] = f2bf(dy); d[2] = f2bf(dz);
            const u16* f = F + (size_t)n * 128;
            for (int c = 0; c < 128; ++c) d[3 + c] = f[c];
            for (int c = 131; c < 160; ++c) d[c] = 0;
        }
        if (first < 0 && mask) first = b0 + __ffs(mask) - 1;
        count += __popc(mask);
    }
    int filled = count < NS ? count : NS;
    int fi = first >= 0 ? first : 0;
    float dx = X[fi * 3 + 0] - cx;
    float dy = X[fi * 3 + 1] - cy;
    float dz = X[fi * 3 + 2] - cz;
    const u16* f = F + (size_t)fi * 128;
    for (int pos = filled + lane; pos < NS; pos += 32) {
        u16* d = base + (size_t)pos * CP;
        d[0] = f2bf(dx); d[1] = f2bf(dy); d[2] = f2bf(dz);
        for (int c = 0; c < 128; ++c) d[3 + c] = f[c];
        for (int c = 131; c < 160; ++c) d[c] = 0;
    }
}

// ---------------------------------------------------------------------------
// SA3 grouping (group-all): rows (B*32, 288) = [xyz2(3), feat2(256), pad(29)]
// ---------------------------------------------------------------------------
__global__ void group3_kernel(const float* __restrict__ xyz2,  // (B*32,3)
                              const u16* __restrict__ feat2,   // (B*32,256)
                              u16* __restrict__ out) {
    int i = blockIdx.x * blockDim.x + threadIdx.x;
    if (i >= 2048) return;
    u16* d = out + (size_t)i * 288;
    d[0] = f2bf(xyz2[(size_t)i * 3 + 0]);
    d[1] = f2bf(xyz2[(size_t)i * 3 + 1]);
    d[2] = f2bf(xyz2[(size_t)i * 3 + 2]);
    const u16* f = feat2 + (size_t)i * 256;
    for (int c = 0; c < 256; ++c) d[3 + c] = f[c];
    for (int c = 259; c < 288; ++c) d[c] = 0;
}

// ---------------------------------------------------------------------------
// WMMA GEMM: D(bf16)[M,N] = A(bf16)[M,Kpad] x Wsw(bf16, pre-swizzled tiles);
// accumulates per-channel sum/sumsq into stats[0..N)/stats[N..2N) for BN.
// Block = 256 threads = 8 waves; block tile 128(M) x 64(N); K-step 32.
// Staging = raw 2KB copy per K-step (1 b64 load + 1 ds_store_b64 per thread).
// A fragment: two b128 global loads/lane (contiguous K: half h holds K 16h..16h+15).
// Requires: M % 16 == 0, N % 64 == 0, Kpad % 32 == 0 (all call sites comply).
// ---------------------------------------------------------------------------
__device__ __forceinline__ void gemm_epilog(const v8f& acc, int ct,
                                            int n0, int r, int m0, int half,
                                            bool rowok, int N, u16* D,
                                            float* ssum, float* ssq) {
    float ls = 0.f, lsq = 0.f;
    int cg = n0 + ct * 16 + r;
    if (rowok) {
        unsigned rbase = (unsigned)(m0 + half * 8);
#pragma unroll
        for (int j = 0; j < 8; ++j) {
            float v = acc[j];
            D[(size_t)(rbase + j) * N + cg] = f2bf(v);
            ls += v; lsq += v * v;
        }
    }
    atomicAdd(&ssum[ct * 16 + r], ls);
    atomicAdd(&ssq [ct * 16 + r], lsq);
}

__global__ __launch_bounds__(256)
void gemm_wmma_bf16(const u16* __restrict__ A, const u16* __restrict__ Wsw,
                    u16* __restrict__ D, float* __restrict__ stats,
                    int M, int N, int Kpad) {
    __shared__ __align__(16) u16 Blds[2048];   // one pre-swizzled 32x64 tile
    __shared__ float ssum[64];
    __shared__ float ssq[64];
    const int tid  = threadIdx.x;
    const int wave = tid >> 5;
    const int lane = tid & 31;
    const int half = lane >> 4;
    const int r    = lane & 15;
    const int m0   = blockIdx.x * 128 + wave * 16;
    const int n0   = blockIdx.y * 64;
    const int row  = m0 + r;
    const bool rowok = row < M;                 // uniform per wave (M % 16 == 0)
    const unsigned arow = (unsigned)row * (unsigned)Kpad;

    // tile stream for this n-block: advance by (N/64)*2KB per K-step
    const uint2* wt = (const uint2*)(Wsw + (size_t)blockIdx.y * 2048) + tid;
    const unsigned wstep = (unsigned)(N >> 6) * 256u;   // uint2 units per K-step

    if (tid < 64) { ssum[tid] = 0.f; ssq[tid] = 0.f; }

    v8f acc0, acc1, acc2, acc3;
#pragma unroll
    for (int i = 0; i < 8; ++i) { acc0[i] = 0.f; acc1[i] = 0.f; acc2[i] = 0.f; acc3[i] = 0.f; }

    for (int k0 = 0; k0 < Kpad; k0 += 32, wt += wstep) {
        __syncthreads();
        ((uint2*)Blds)[tid] = *wt;              // stage pre-swizzled tile
        __syncthreads();

        Frag a;
        if (rowok) {
            const uint4* p = (const uint4*)(A + arow + (unsigned)k0);
            a.q[0] = p[2 * half];    // K = k0 + half*16 .. +7
            a.q[1] = p[2 * half + 1];    // K = k0 + half*16 + 8 .. +15
            if (k0 + 32 < Kpad) __builtin_prefetch(A + arow + (unsigned)k0 + 32u, 0, 1);
        } else {
            a.q[0] = make_uint4(0, 0, 0, 0);
            a.q[1] = make_uint4(0, 0, 0, 0);
        }

        const uint4* bl = (const uint4*)Blds;
        Frag b;
        b.q[0] = bl[(0 * 32 + lane) * 2 + 0]; b.q[1] = bl[(0 * 32 + lane) * 2 + 1];
        acc0 = __builtin_amdgcn_wmma_f32_16x16x32_bf16(false, a.v, false, b.v, (short)0, acc0, false, false);
        b.q[0] = bl[(1 * 32 + lane) * 2 + 0]; b.q[1] = bl[(1 * 32 + lane) * 2 + 1];
        acc1 = __builtin_amdgcn_wmma_f32_16x16x32_bf16(false, a.v, false, b.v, (short)0, acc1, false, false);
        b.q[0] = bl[(2 * 32 + lane) * 2 + 0]; b.q[1] = bl[(2 * 32 + lane) * 2 + 1];
        acc2 = __builtin_amdgcn_wmma_f32_16x16x32_bf16(false, a.v, false, b.v, (short)0, acc2, false, false);
        b.q[0] = bl[(3 * 32 + lane) * 2 + 0]; b.q[1] = bl[(3 * 32 + lane) * 2 + 1];
        acc3 = __builtin_amdgcn_wmma_f32_16x16x32_bf16(false, a.v, false, b.v, (short)0, acc3, false, false);
    }

    gemm_epilog(acc0, 0, n0, r, m0, half, rowok, N, D, ssum, ssq);
    gemm_epilog(acc1, 1, n0, r, m0, half, rowok, N, D, ssum, ssq);
    gemm_epilog(acc2, 2, n0, r, m0, half, rowok, N, D, ssum, ssq);
    gemm_epilog(acc3, 3, n0, r, m0, half, rowok, N, D, ssum, ssq);
    __syncthreads();
    if (tid < 64) {
        atomicAdd(&stats[n0 + tid],     ssum[tid]);
        atomicAdd(&stats[N + n0 + tid], ssq[tid]);
    }
}

// ---------------------------------------------------------------------------
// BatchNorm (global per-channel over all M rows) + ReLU, in place on bf16.
// ---------------------------------------------------------------------------
__global__ void bn_relu_bf16(u16* __restrict__ h, const float* __restrict__ stats,
                             const float* __restrict__ g, const float* __restrict__ b,
                             int M, int N) {
    size_t i = (size_t)blockIdx.x * blockDim.x + threadIdx.x;
    size_t total = (size_t)M * N;
    if (i >= total) return;
    int c = (int)(i % N);
    float cnt = (float)M;
    float mu  = stats[c] / cnt;
    float var = stats[N + c] / cnt - mu * mu;
    float inv = rsqrtf(var + BN_EPS);
    float sc  = g[c] * inv;
    float sh  = b[c] - mu * sc;
    float v = bf2f(h[i]) * sc + sh;
    h[i] = f2bf(v > 0.f ? v : 0.f);
}

// ---------------------------------------------------------------------------
// Max-pool over group dim K: in (BS,K,N) bf16 -> out (BS,N) bf16
// ---------------------------------------------------------------------------
__global__ void maxpool_kernel(const u16* __restrict__ in, u16* __restrict__ out,
                               int BS, int K, int N) {
    int i = blockIdx.x * blockDim.x + threadIdx.x;
    if (i >= BS * N) return;
    int bs = i / N, c = i % N;
    float m = -3.0e38f;
    const u16* p = in + (size_t)bs * K * N + c;
    for (int k = 0; k < K; ++k) {
        float v = bf2f(p[(size_t)k * N]);
        m = v > m ? v : m;
    }
    out[(size_t)bs * N + c] = f2bf(m);
}

// ---------------------------------------------------------------------------
// Head: out[row] = x[row,:] . hw + hb     (64 x 1024 -> 64)
// ---------------------------------------------------------------------------
__global__ __launch_bounds__(256)
void head_kernel(const u16* __restrict__ x, const float* __restrict__ hw,
                 const float* __restrict__ hb, float* __restrict__ out) {
    __shared__ float red[256];
    int row = blockIdx.x, tid = threadIdx.x;
    float s = 0.f;
    for (int k = tid; k < 1024; k += 256)
        s += bf2f(x[(size_t)row * 1024 + k]) * hw[k];
    red[tid] = s;
    __syncthreads();
    for (int st = 128; st > 0; st >>= 1) {
        if (tid < st) red[tid] += red[tid + st];
        __syncthreads();
    }
    if (tid == 0) out[row] = red[0] + hb[0];
}

__global__ void zero_f32(float* __restrict__ p, int n) {
    int i = blockIdx.x * blockDim.x + threadIdx.x;
    if (i < n) p[i] = 0.f;
}

// ---------------------------------------------------------------------------
extern "C" void kernel_launch(void* const* d_in, const int* in_sizes, int n_in,
                              void* d_out, int out_size, void* d_ws, size_t ws_size,
                              hipStream_t stream) {
    (void)in_sizes; (void)n_in; (void)out_size; (void)ws_size;
    const float* pc  = (const float*)d_in[0];
    const float* gp  = (const float*)d_in[1];
    const float* w1a = (const float*)d_in[2],  *g1a = (const float*)d_in[3],  *b1a = (const float*)d_in[4];
    const float* w1b = (const float*)d_in[5],  *g1b = (const float*)d_in[6],  *b1b = (const float*)d_in[7];
    const float* w1c = (const float*)d_in[8],  *g1c = (const float*)d_in[9],  *b1c = (const float*)d_in[10];
    const float* w2a = (const float*)d_in[11], *g2a = (const float*)d_in[12], *b2a = (const float*)d_in[13];
    const float* w2b = (const float*)d_in[14], *g2b = (const float*)d_in[15], *b2b = (const float*)d_in[16];
    const float* w2c = (const float*)d_in[17], *g2c = (const float*)d_in[18], *b2c = (const float*)d_in[19];
    const float* w3a = (const float*)d_in[20], *g3a = (const float*)d_in[21], *b3a = (const float*)d_in[22];
    const float* w3b = (const float*)d_in[23], *g3b = (const float*)d_in[24], *b3b = (const float*)d_in[25];
    const float* w3c = (const float*)d_in[26], *g3c = (const float*)d_in[27], *b3c = (const float*)d_in[28];
    const float* fw1 = (const float*)d_in[29];                     // fb1 (d_in[30]) cancels in BN
    const float* fg1 = (const float*)d_in[31], *fbb1 = (const float*)d_in[32];
    const float* fw2 = (const float*)d_in[33];                     // fb2 (d_in[34]) cancels in BN
    const float* fg2 = (const float*)d_in[35], *fbb2 = (const float*)d_in[36];
    const float* hw  = (const float*)d_in[37], *hb  = (const float*)d_in[38];

    char* ws = (char*)d_ws;
    size_t off = 0;
    auto alloc = [&](size_t bytes) -> char* {
        char* p = ws + off;
        off = (off + bytes + 255) & ~(size_t)255;
        return p;
    };
    float* xyz0  = (float*)alloc(64ull * 1280 * 3 * 4);
    float* nxyz1 = (float*)alloc(64ull * 128 * 3 * 4);
    float* nxyz2 = (float*)alloc(64ull * 32 * 3 * 4);
    u16*   grp1  = (u16*)  alloc(524288ull * 32 * 2);
    u16*   grp2  = (u16*)  alloc(262144ull * 160 * 2);
    u16*   grp3  = (u16*)  alloc(2048ull * 288 * 2);
    u16*   bufA  = (u16*)  alloc(524288ull * 128 * 2);
    u16*   bufB  = (u16*)  alloc(524288ull * 128 * 2);
    u16*   pool1 = (u16*)  alloc(8192ull * 128 * 2);
    u16*   pool2 = (u16*)  alloc(2048ull * 256 * 2);
    u16*   fcx   = (u16*)  alloc(64ull * 512 * 2);
    float* stats = (float*)alloc(2048ull * 4);
    // pre-swizzled bf16 weights (Kpad x N each)
    u16* sw1a = (u16*)alloc(32ull   * 64   * 2);
    u16* sw1b = (u16*)alloc(64ull   * 64   * 2);
    u16* sw1c = (u16*)alloc(64ull   * 128  * 2);
    u16* sw2a = (u16*)alloc(160ull  * 128  * 2);
    u16* sw2b = (u16*)alloc(128ull  * 128  * 2);
    u16* sw2c = (u16*)alloc(128ull  * 256  * 2);
    u16* sw3a = (u16*)alloc(288ull  * 256  * 2);
    u16* sw3b = (u16*)alloc(256ull  * 256  * 2);
    u16* sw3c = (u16*)alloc(256ull  * 512  * 2);
    u16* swf1 = (u16*)alloc(512ull  * 1024 * 2);
    u16* swf2 = (u16*)alloc(1024ull * 1024 * 2);

    auto prep = [&](const float* W, u16* Wsw, int N, int Kreal, int Kpad) {
        int tot = Kpad * N;
        prep_w_kernel<<<(tot + 255) / 256, 256, 0, stream>>>(W, Wsw, N, Kreal, Kpad);
    };
    auto gemm = [&](const u16* A, const u16* Wsw, u16* Dst, int M, int N, int Kpad) {
        zero_f32<<<(2 * N + 255) / 256, 256, 0, stream>>>(stats, 2 * N);
        dim3 grid((M + 127) / 128, N / 64);
        gemm_wmma_bf16<<<grid, 256, 0, stream>>>(A, Wsw, Dst, stats, M, N, Kpad);
    };
    auto bn = [&](u16* h, const float* g, const float* b, int M, int N) {
        size_t tot = (size_t)M * N;
        bn_relu_bf16<<<(unsigned)((tot + 255) / 256), 256, 0, stream>>>(h, stats, g, b, M, N);
    };

    // ---- one-time weight conversion / swizzle ----
    prep(w1a, sw1a, 64,   7,    32);
    prep(w1b, sw1b, 64,   64,   64);
    prep(w1c, sw1c, 128,  64,   64);
    prep(w2a, sw2a, 128,  131,  160);
    prep(w2b, sw2b, 128,  128,  128);
    prep(w2c, sw2c, 256,  128,  128);
    prep(w3a, sw3a, 256,  259,  288);
    prep(w3b, sw3b, 256,  256,  256);
    prep(w3c, sw3c, 512,  256,  256);
    prep(fw1, swf1, 1024, 512,  512);
    prep(fw2, swf2, 1024, 1024, 1024);

    // ---- input assembly ----
    build_xyz_kernel<<<(64 * 1280 * 3 + 255) / 256, 256, 0, stream>>>(pc, gp, xyz0, 64, 1024, 256);

    // ---- SA1 : 128 centers, 64 neighbors, 7->64->64->128 ----
    fps_kernel<<<64, 256, 0, stream>>>(xyz0, nxyz1, 1280, 128);
    group1_kernel<<<1024, 256, 0, stream>>>(xyz0, nxyz1, grp1);
    gemm(grp1, sw1a, bufA, 524288, 64, 32);    bn(bufA, g1a, b1a, 524288, 64);
    gemm(bufA, sw1b, bufB, 524288, 64, 64);    bn(bufB, g1b, b1b, 524288, 64);
    gemm(bufB, sw1c, bufA, 524288, 128, 64);   bn(bufA, g1c, b1c, 524288, 128);
    maxpool_kernel<<<(8192 * 128 + 255) / 256, 256, 0, stream>>>(bufA, pool1, 8192, 64, 128);

    // ---- SA2 : 32 centers, 128 neighbors, 131->128->128->256 ----
    fps_kernel<<<64, 256, 0, stream>>>(nxyz1, nxyz2, 128, 32);
    group2_kernel<<<256, 256, 0, stream>>>(nxyz1, nxyz2, pool1, grp2);
    gemm(grp2, sw2a, bufA, 262144, 128, 160);  bn(bufA, g2a, b2a, 262144, 128);
    gemm(bufA, sw2b, bufB, 262144, 128, 128);  bn(bufB, g2b, b2b, 262144, 128);
    gemm(bufB, sw2c, bufA, 262144, 256, 128);  bn(bufA, g2c, b2c, 262144, 256);
    maxpool_kernel<<<(2048 * 256 + 255) / 256, 256, 0, stream>>>(bufA, pool2, 2048, 128, 256);

    // ---- SA3 : group-all, 259->256->256->512 ----
    group3_kernel<<<8, 256, 0, stream>>>(nxyz2, pool2, grp3);
    gemm(grp3, sw3a, bufA, 2048, 256, 288);    bn(bufA, g3a, b3a, 2048, 256);
    gemm(bufA, sw3b, bufB, 2048, 256, 256);    bn(bufB, g3b, b3b, 2048, 256);
    gemm(bufB, sw3c, bufA, 2048, 512, 256);    bn(bufA, g3c, b3c, 2048, 512);
    maxpool_kernel<<<(64 * 512 + 255) / 256, 256, 0, stream>>>(bufA, fcx, 64, 32, 512);

    // ---- FC head : 512->1024->1024->1 ----
    gemm(fcx,  swf1, bufB, 64, 1024, 512);     bn(bufB, fg1, fbb1, 64, 1024);
    gemm(bufB, swf2, bufA, 64, 1024, 1024);    bn(bufA, fg2, fbb2, 64, 1024);
    head_kernel<<<64, 256, 0, stream>>>(bufA, hw, hb, (float*)d_out);
}
